// NeuralFBSDE_52269751992465
// MI455X (gfx1250) — matrix operand order repaired and checked by reading, not compiled, not run on hardware
//
#include <hip/hip_runtime.h>

typedef __attribute__((ext_vector_type(16))) _Float16 v16h;
typedef __attribute__((ext_vector_type(8)))  _Float16 v8h;
typedef __attribute__((ext_vector_type(8)))  float    v8f;

#define WIDTH  256
#define HDIM   100
#define INDIM  101
#define INPAD  128
#define MTILE  32     // trajectories (rows) per block
#define NWAVES 8      // 256 threads / wave32

// f16 workspace layout (element offsets)
#define O_W0F 0        // [256][128] forward  (row n=out, contiguous k=in, zero-padded 101..127)
#define O_W0T 32768    // [128][256] backward (row j=in,  contiguous k=out)
#define O_W1F 65536
#define O_W1T 131072
#define O_W2F 196608
#define O_W2T 262144
#define O_W3F 327680
#define O_W3T 393216
#define O_W4F 458752   // [256]
#define WS_F16_TOTAL 459008

// ---------------- RNG: hash + Box-Muller ----------------
__device__ __forceinline__ unsigned hashu(unsigned x) {
  x ^= x >> 17; x *= 0xed5ad4bbu;
  x ^= x >> 11; x *= 0xac4c1b51u;
  x ^= x >> 15; x *= 0x31848babu;
  x ^= x >> 14;
  return x;
}
__device__ __forceinline__ float gaussf(unsigned seed) {
  unsigned a = hashu(seed);
  unsigned b = hashu(seed ^ 0x9e3779b9u);
  float u1 = ((a >> 8) + 1u) * (1.0f / 16777217.0f);
  float u2 = (b >> 8) * (1.0f / 16777216.0f);
  float r = __builtin_sqrtf(-2.0f * __logf(u1));
  return r * __cosf(6.28318530718f * u2);
}

// ---------------- WMMA fragment loaders ----------------
// A (16x32 f16) from LDS, row-major stride WIDTH.
// lanes 0-15: K = {0..15}; lanes 16-31: K = {16..31}
__device__ __forceinline__ v16h load_a(const _Float16* actIn, int r, int kc, int kh) {
  const _Float16* p = actIn + r * WIDTH + kc * 32 + kh * 16;
  v8h lo = *(const v8h*)p;
  v8h hi = *(const v8h*)(p + 8);
  v16h a;
#pragma unroll
  for (int i = 0; i < 8; ++i) { a[i] = lo[i]; a[i + 8] = hi[i]; }
  return a;
}

#define WMMA_F16(a, b, c) \
  __builtin_amdgcn_wmma_f32_16x16x32_f16(false, (a), false, (b), (short)0, (c), false, false)

// ---------------- GEMM layers (2x2 register-blocked per wave) ----------------
// Forward: out[m][n] = relu( sum_k in[m][k] * Wf[n][k] + bias[n] ), M=32, N=256.
// Wave w owns column tiles nt0=w, nt1=w+8 and BOTH row tiles; A frags shared
// across columns, B frags shared across rows -> 4 WMMA per (a0,a1,b0,b1) load.
__device__ __forceinline__ void layer_fwd(const _Float16* __restrict__ Wf,
                                          const float* __restrict__ bias,
                                          const _Float16* actIn, _Float16* actOut, int K) {
  const int tid  = threadIdx.x;
  const int wave = tid >> 5;
  const int lane = tid & 31;
  const int l15  = lane & 15;
  const int kh   = lane >> 4;
  const int KC   = K >> 5;
  const int n0 = wave * 16 + l15;
  const int n1 = (wave + 8) * 16 + l15;
  const _Float16* w0 = Wf + (size_t)n0 * K;
  const _Float16* w1 = Wf + (size_t)n1 * K;
  v8f acc00 = {0.f,0.f,0.f,0.f,0.f,0.f,0.f,0.f};
  v8f acc01 = acc00, acc10 = acc00, acc11 = acc00;
  for (int kc = 0; kc < KC; ++kc) {
    v16h a0 = load_a(actIn, l15,      kc, kh);   // rows 0..15
    v16h a1 = load_a(actIn, 16 + l15, kc, kh);   // rows 16..31
    v16h b0 = *(const v16h*)(w0 + kc * 32 + kh * 16);
    v16h b1 = *(const v16h*)(w1 + kc * 32 + kh * 16);
    acc00 = WMMA_F16(a0, b0, acc00);
    acc01 = WMMA_F16(a0, b1, acc01);
    acc10 = WMMA_F16(a1, b0, acc10);
    acc11 = WMMA_F16(a1, b1, acc11);
  }
  const float bn0 = bias[n0];
  const float bn1 = bias[n1];
#pragma unroll
  for (int v = 0; v < 8; ++v) {
    const int r0 = kh * 8 + v;
    const int r1 = 16 + kh * 8 + v;
    float v00 = acc00[v] + bn0; v00 = v00 > 0.f ? v00 : 0.f;
    float v01 = acc01[v] + bn1; v01 = v01 > 0.f ? v01 : 0.f;
    float v10 = acc10[v] + bn0; v10 = v10 > 0.f ? v10 : 0.f;
    float v11 = acc11[v] + bn1; v11 = v11 > 0.f ? v11 : 0.f;
    actOut[r0 * WIDTH + n0] = (_Float16)v00;
    actOut[r0 * WIDTH + n1] = (_Float16)v01;
    actOut[r1 * WIDTH + n0] = (_Float16)v10;
    actOut[r1 * WIDTH + n1] = (_Float16)v11;
  }
}

// Backward: g_out[m][n] = mask[m][n] * sum_k g_in[m][k] * WT[n][k]
__device__ __forceinline__ void layer_bwd(const _Float16* __restrict__ WT,
                                          const unsigned short* __restrict__ maskL,
                                          const _Float16* gIn, _Float16* gOut) {
  const int tid  = threadIdx.x;
  const int wave = tid >> 5;
  const int lane = tid & 31;
  const int l15  = lane & 15;
  const int kh   = lane >> 4;
  const int nt0 = wave, nt1 = wave + 8;
  const int n0 = nt0 * 16 + l15;
  const int n1 = nt1 * 16 + l15;
  const _Float16* w0 = WT + (size_t)n0 * WIDTH;
  const _Float16* w1 = WT + (size_t)n1 * WIDTH;
  v8f acc00 = {0.f,0.f,0.f,0.f,0.f,0.f,0.f,0.f};
  v8f acc01 = acc00, acc10 = acc00, acc11 = acc00;
  for (int kc = 0; kc < WIDTH / 32; ++kc) {
    v16h a0 = load_a(gIn, l15,      kc, kh);
    v16h a1 = load_a(gIn, 16 + l15, kc, kh);
    v16h b0 = *(const v16h*)(w0 + kc * 32 + kh * 16);
    v16h b1 = *(const v16h*)(w1 + kc * 32 + kh * 16);
    acc00 = WMMA_F16(a0, b0, acc00);
    acc01 = WMMA_F16(a0, b1, acc01);
    acc10 = WMMA_F16(a1, b0, acc10);
    acc11 = WMMA_F16(a1, b1, acc11);
  }
#pragma unroll
  for (int v = 0; v < 8; ++v) {
    const int r0 = kh * 8 + v;
    const int r1 = 16 + kh * 8 + v;
    const unsigned short m00 = maskL[r0 * 16 + nt0];
    const unsigned short m01 = maskL[r0 * 16 + nt1];
    const unsigned short m10 = maskL[r1 * 16 + nt0];
    const unsigned short m11 = maskL[r1 * 16 + nt1];
    gOut[r0 * WIDTH + n0] = (_Float16)(((m00 >> l15) & 1) ? acc00[v] : 0.f);
    gOut[r0 * WIDTH + n1] = (_Float16)(((m01 >> l15) & 1) ? acc01[v] : 0.f);
    gOut[r1 * WIDTH + n0] = (_Float16)(((m10 >> l15) & 1) ? acc10[v] : 0.f);
    gOut[r1 * WIDTH + n1] = (_Float16)(((m11 >> l15) & 1) ? acc11[v] : 0.f);
  }
}

// du/dx: z[m][j] = sum_k g1[m][k] * W0T[j][k], keep columns 1..100.
// Wave w owns column tile nt=w (8 tiles), both row tiles share the B fragment.
__device__ __forceinline__ void layer_dudx(const _Float16* __restrict__ W0T,
                                           const _Float16* gIn, float* sZ) {
  const int tid  = threadIdx.x;
  const int wave = tid >> 5;
  const int lane = tid & 31;
  const int l15  = lane & 15;
  const int kh   = lane >> 4;
  const int n = wave * 16 + l15;
  const _Float16* wrow = W0T + (size_t)n * WIDTH;
  v8f acc0 = {0.f,0.f,0.f,0.f,0.f,0.f,0.f,0.f};
  v8f acc1 = acc0;
  for (int kc = 0; kc < WIDTH / 32; ++kc) {
    v16h b  = *(const v16h*)(wrow + kc * 32 + kh * 16);
    v16h a0 = load_a(gIn, l15,      kc, kh);
    v16h a1 = load_a(gIn, 16 + l15, kc, kh);
    acc0 = WMMA_F16(a0, b, acc0);
    acc1 = WMMA_F16(a1, b, acc1);
  }
  if (n >= 1 && n <= HDIM) {
#pragma unroll
    for (int v = 0; v < 8; ++v) {
      const int r0 = kh * 8 + v;
      const int r1 = 16 + kh * 8 + v;
      sZ[r0 * HDIM + (n - 1)] = acc0[v];
      sZ[r1 * HDIM + (n - 1)] = acc1[v];
    }
  }
}

__device__ __forceinline__ void pack_mask(const _Float16* act, unsigned short* maskL) {
  for (int idx = threadIdx.x; idx < MTILE * 16; idx += 256) {
    const int r = idx >> 4, nt = idx & 15;
    const _Float16* p = act + r * WIDTH + nt * 16;
    unsigned short m = 0;
#pragma unroll
    for (int i = 0; i < 16; ++i)
      if ((float)p[i] > 0.f) m |= (unsigned short)(1u << i);
    maskL[idx] = m;
  }
}

// ---------------- full MLP forward + VJP ----------------
__device__ __forceinline__ void eval_mlp(
    const _Float16* __restrict__ ws,
    const float* __restrict__ b0, const float* __restrict__ b1,
    const float* __restrict__ b2, const float* __restrict__ b3,
    const float* __restrict__ b4,
    float tcur, const float* sX, float* sZ, float* sY,
    _Float16* sA, _Float16* sB, unsigned short (*sMask)[MTILE * 16], float* sRed,
    float* outYs, int rowBase) {
  const int tid = threadIdx.x;
  // build input tile [t, x, 0-pad] into sA (cols 0..127, stride 256)
  for (int idx = tid; idx < MTILE * INPAD; idx += 256) {
    const int r = idx >> 7, c = idx & 127;
    float v = (c == 0) ? tcur : ((c <= HDIM) ? sX[r * HDIM + (c - 1)] : 0.f);
    sA[r * WIDTH + c] = (_Float16)v;
  }
  __syncthreads();
  layer_fwd(ws + O_W0F, b0, sA, sB, INPAD);  __syncthreads();
  pack_mask(sB, sMask[0]);                   __syncthreads();
  layer_fwd(ws + O_W1F, b1, sB, sA, WIDTH);  __syncthreads();
  pack_mask(sA, sMask[1]);                   __syncthreads();
  layer_fwd(ws + O_W2F, b2, sA, sB, WIDTH);  __syncthreads();
  pack_mask(sB, sMask[2]);                   __syncthreads();
  layer_fwd(ws + O_W3F, b3, sB, sA, WIDTH);  __syncthreads();
  pack_mask(sA, sMask[3]);                   __syncthreads();

  // y = h4 @ W4^T + b4  (h4 in sA)
  {
    const int row = tid >> 3, l8 = tid & 7;
    const _Float16* w4 = ws + O_W4F;
    const _Float16* h  = sA + row * WIDTH;
    float s = 0.f;
    for (int k = l8 * 32; k < l8 * 32 + 32; ++k) s += (float)h[k] * (float)w4[k];
    sRed[row * 16 + l8] = s;
  }
  __syncthreads();
  if (tid < MTILE) {
    float y = 0.f;
    for (int i = 0; i < 8; ++i) y += sRed[tid * 16 + i];
    y += b4[0];
    sY[tid] = y;
    if (outYs) outYs[rowBase + tid] = y;
  }
  __syncthreads();

  // g4 = mask4 ? W4 : 0   into sB
  {
    const _Float16* w4 = ws + O_W4F;
    for (int idx = tid; idx < MTILE * WIDTH; idx += 256) {
      const int r = idx >> 8, c = idx & 255;
      const unsigned short mw = sMask[3][r * 16 + (c >> 4)];
      sB[idx] = ((mw >> (c & 15)) & 1) ? w4[c] : (_Float16)0.f;
    }
  }
  __syncthreads();
  layer_bwd(ws + O_W3T, sMask[2], sB, sA); __syncthreads();
  layer_bwd(ws + O_W2T, sMask[1], sA, sB); __syncthreads();
  layer_bwd(ws + O_W1T, sMask[0], sB, sA); __syncthreads();
  layer_dudx(ws + O_W0T, sA, sZ);          __syncthreads();
}

// ---------------- weight precast kernel (f32 -> f16, +transposes) ----------------
__global__ __launch_bounds__(256) void prep_kernel(
    const float* __restrict__ W0, const float* __restrict__ W1, const float* __restrict__ W2,
    const float* __restrict__ W3, const float* __restrict__ W4, _Float16* __restrict__ ws) {
  int idx = blockIdx.x * blockDim.x + threadIdx.x;
  if (idx < 32768) { // W0f [256][128]
    const int n = idx >> 7, k = idx & 127;
    ws[O_W0F + idx] = (_Float16)((k < INDIM) ? W0[n * INDIM + k] : 0.f);
    return;
  }
  idx -= 32768;
  if (idx < 32768) { // W0T [128][256]: W0T[j][k] = W0[k][j]
    const int j = idx >> 8, k = idx & 255;
    ws[O_W0T + idx] = (_Float16)((j < INDIM) ? W0[k * INDIM + j] : 0.f);
    return;
  }
  idx -= 32768;
  if (idx < 65536) { ws[O_W1F + idx] = (_Float16)W1[idx]; return; }
  idx -= 65536;
  if (idx < 65536) { const int j = idx >> 8, k = idx & 255; ws[O_W1T + idx] = (_Float16)W1[k * WIDTH + j]; return; }
  idx -= 65536;
  if (idx < 65536) { ws[O_W2F + idx] = (_Float16)W2[idx]; return; }
  idx -= 65536;
  if (idx < 65536) { const int j = idx >> 8, k = idx & 255; ws[O_W2T + idx] = (_Float16)W2[k * WIDTH + j]; return; }
  idx -= 65536;
  if (idx < 65536) { ws[O_W3F + idx] = (_Float16)W3[idx]; return; }
  idx -= 65536;
  if (idx < 65536) { const int j = idx >> 8, k = idx & 255; ws[O_W3T + idx] = (_Float16)W3[k * WIDTH + j]; return; }
  idx -= 65536;
  if (idx < 256) { ws[O_W4F + idx] = (_Float16)W4[idx]; return; }
}

// ---------------- persistent FBSDE kernel: T-step loop, block-local state ----------------
__global__ __launch_bounds__(256) void fbsde_kernel(
    const float* __restrict__ x0, const float* __restrict__ t0p, const float* __restrict__ dtp,
    const int T, const int B,
    const float* __restrict__ b0, const float* __restrict__ b1, const float* __restrict__ b2,
    const float* __restrict__ b3, const float* __restrict__ b4,
    const _Float16* __restrict__ ws,
    float* __restrict__ outX, float* __restrict__ outY, float* __restrict__ outZ,
    float* __restrict__ outXs, float* __restrict__ outYt, float* __restrict__ outYs) {
  __shared__ __align__(32) _Float16 sA[MTILE * WIDTH];
  __shared__ __align__(32) _Float16 sB[MTILE * WIDTH];
  __shared__ float sX[MTILE * HDIM];
  __shared__ float sZ[MTILE * HDIM];
  __shared__ float sY[MTILE];
  __shared__ unsigned short sMask[4][MTILE * 16];
  __shared__ float sRed[MTILE * 16];

  const int tid = threadIdx.x;
  const int rowBase = blockIdx.x * MTILE;
  const float t0v = t0p[0];
  const float dtv = dtp[0];
  const float sqdt = __builtin_sqrtf(dtv);

  // load x0 tile into LDS (persists across all T steps)
  for (int idx = tid; idx < MTILE * HDIM; idx += 256) {
    const int r = idx / HDIM, c = idx - r * HDIM;
    sX[idx] = x0[(size_t)(rowBase + r) * HDIM + c];
  }
  __syncthreads();

  // initial eval at t = 0 -> y0, z0
  eval_mlp(ws, b0, b1, b2, b3, b4, 0.0f, sX, sZ, sY, sA, sB, sMask, sRed, nullptr, 0);

  for (int t = 0; t < T; ++t) {
    // --- Euler–Maruyama step + y_tilde ---
    {
      const int row = tid >> 3, l8 = tid & 7;
      float s1 = 0.f, s2 = 0.f;
      for (int j = l8; j < HDIM; j += 8) {
        const float xo = sX[row * HDIM + j];
        const float z  = sZ[row * HDIM + j];
        const unsigned seed =
            ((unsigned)t * (unsigned)B + (unsigned)(rowBase + row)) * 128u + (unsigned)j;
        const float dW = gaussf(seed) * sqdt;
        const float sig = 0.4f * xo;        // sigma_fn
        const float xn = xo + sig * dW;     // mu_fn == 0
        s1 += xo * z;
        s2 += z * sig * dW;
        sX[row * HDIM + j] = xn;
        outXs[(size_t)t * B * HDIM + (size_t)(rowBase + row) * HDIM + j] = xn;
      }
      sRed[row * 16 + l8]     = s1;
      sRed[row * 16 + 8 + l8] = s2;
    }
    __syncthreads();
    if (tid < MTILE) {
      float S1 = 0.f, S2 = 0.f;
      for (int i = 0; i < 8; ++i) { S1 += sRed[tid * 16 + i]; S2 += sRed[tid * 16 + 8 + i]; }
      const float y = sY[tid];
      const float yt = y + 0.05f * (y - S1) * dtv + S2;   // phi*dt + z.sig.dW
      outYt[(size_t)t * B + rowBase + tid] = yt;
    }
    __syncthreads();

    const float tnext = t0v + (float)(t + 1) * dtv;
    eval_mlp(ws, b0, b1, b2, b3, b4, tnext, sX, sZ, sY, sA, sB, sMask, sRed,
             outYs + (size_t)t * B, rowBase);
  }

  // final carries (x, y, z)
  for (int idx = tid; idx < MTILE * HDIM; idx += 256) {
    const int r = idx / HDIM, c = idx - r * HDIM;
    outX[(size_t)(rowBase + r) * HDIM + c] = sX[idx];
    outZ[(size_t)(rowBase + r) * HDIM + c] = sZ[idx];
  }
  if (tid < MTILE) outY[rowBase + tid] = sY[tid];
}

extern "C" void kernel_launch(void* const* d_in, const int* in_sizes, int n_in,
                              void* d_out, int out_size, void* d_ws, size_t ws_size,
                              hipStream_t stream) {
  (void)n_in; (void)ws_size;
  const float* x0  = (const float*)d_in[0];
  const float* t0  = (const float*)d_in[1];
  const float* dt  = (const float*)d_in[2];
  // d_in[3] = num_timesteps (device int); T is derived from out_size instead (host-visible)
  const float* W0  = (const float*)d_in[4];
  const float* b0  = (const float*)d_in[5];
  const float* W1  = (const float*)d_in[6];
  const float* b1  = (const float*)d_in[7];
  const float* W2  = (const float*)d_in[8];
  const float* b2  = (const float*)d_in[9];
  const float* W3  = (const float*)d_in[10];
  const float* b3  = (const float*)d_in[11];
  const float* W4  = (const float*)d_in[12];
  const float* b4  = (const float*)d_in[13];

  const int IN = in_sizes[4] / WIDTH;   // 101
  const int H  = IN - 1;                // 100
  const int B  = in_sizes[0] / H;       // 4096
  // out = x(B*H) + y(B) + z(B*H) + xs(T*B*H) + yt(T*B) + ys(T*B)
  const long long carry = 2LL * B * H + B;
  const long long T = ((long long)out_size - carry) / ((long long)B * H + 2LL * B);

  _Float16* wsh = (_Float16*)d_ws;
  prep_kernel<<<(WS_F16_TOTAL + 255) / 256, 256, 0, stream>>>(W0, W1, W2, W3, W4, wsh);

  float* out    = (float*)d_out;
  float* out_x  = out;
  float* out_y  = out_x + (size_t)B * H;
  float* out_z  = out_y + B;
  float* out_xs = out_z + (size_t)B * H;
  float* out_yt = out_xs + (size_t)T * B * H;
  float* out_ys = out_yt + (size_t)T * B;

  fbsde_kernel<<<B / MTILE, 256, 0, stream>>>(
      x0, t0, dt, (int)T, B, b0, b1, b2, b3, b4, wsh,
      out_x, out_y, out_z, out_xs, out_yt, out_ys);
}
